// StationToGrid_68324339745129
// MI455X (gfx1250) — matrix operand order repaired and checked by reading, not compiled, not run on hardware
//
#include <hip/hip_runtime.h>
#include <hip/hip_bf16.h>
#include <math.h>

#define HIDDEN_DIM 256
#define NLAT 64
#define NLON 128
#define GRID_PTS (NLAT * NLON)   // 8192
#define BATCH 2
#define NST 1024                 // stations

typedef _Float16 half_t;
typedef _Float16 v16h __attribute__((ext_vector_type(16)));
typedef float    v8f  __attribute__((ext_vector_type(8)));
typedef float    v4f  __attribute__((ext_vector_type(4)));
typedef unsigned int v4u __attribute__((ext_vector_type(4)));
typedef unsigned int v8u __attribute__((ext_vector_type(8)));

// ---------------------------------------------------------------------------
// Kernel 0: station_features [b][s][d] f32  ->  Ft [b][d][s] f16
// (so the WMMA B-matrix, K-major per lane, loads contiguously)
// ---------------------------------------------------------------------------
__global__ void feat_transpose_kernel(const float* __restrict__ F,
                                      half_t* __restrict__ Ft) {
    int idx = blockIdx.x * blockDim.x + threadIdx.x;
    if (idx >= BATCH * NST * HIDDEN_DIM) return;
    int d = idx % HIDDEN_DIM;
    int s = (idx / HIDDEN_DIM) % NST;
    int b = idx / (HIDDEN_DIM * NST);
    Ft[(size_t)(b * HIDDEN_DIM + d) * NST + s] = (half_t)F[idx];
}

// ---------------------------------------------------------------------------
// Kernel 1: per-(b,g) wave computes the MLP weight for all 1024 stations,
// softmaxes across stations (wave32 shuffle reductions), stores f16 P[b][g][s].
// ---------------------------------------------------------------------------
__global__ void station_weights_kernel(const float* __restrict__ coords,
                                       const float* __restrict__ mask,
                                       const float* __restrict__ W1,
                                       const float* __restrict__ b1,
                                       const float* __restrict__ W2,
                                       const float* __restrict__ b2,
                                       half_t* __restrict__ P) {
    // MLP weights in LDS: [0..95] W1 (3x32), [96..127] b1, [128..159] W2, [160] b2
    __shared__ float lw[161];
    for (int i = threadIdx.x; i < 161; i += blockDim.x) {
        float v;
        if (i < 96)       v = W1[i];
        else if (i < 128) v = b1[i - 96];
        else if (i < 160) v = W2[i - 128];
        else              v = b2[0];
        lw[i] = v;
    }
    __syncthreads();

    const int lane = threadIdx.x & 31;
    const int wave = blockIdx.x * (blockDim.x >> 5) + (threadIdx.x >> 5);
    if (wave >= BATCH * GRID_PTS) return;
    const int b = wave / GRID_PTS;
    const int g = wave % GRID_PTS;

    const float latg = -90.0f  + (float)(g >> 7)  * (180.0f / 63.0f);
    const float long_ = -180.0f + (float)(g & 127) * (360.0f / 127.0f);
    const float bias2 = lw[160];

    float rall[32];  // this lane's 32 stations: s = i*32 + lane

    #pragma unroll
    for (int cb = 0; cb < 4; ++cb) {     // 4 blocks of 8 stations per lane
        float dla[8], dlo[8], dst[8], acc[8], msk[8];
        #pragma unroll
        for (int t = 0; t < 8; ++t) {
            int s = ((cb * 8 + t) << 5) + lane;
            size_t cbase = ((size_t)(b * NST + s)) * 3;
            float a = coords[cbase + 0] - latg;
            float o = coords[cbase + 1] - long_;
            dla[t] = a; dlo[t] = o;
            dst[t] = sqrtf(fmaf(a, a, fmaf(o, o, 1e-6f)));
            acc[t] = bias2;
            msk[t] = mask[b * NST + s];
        }
        for (int j = 0; j < 32; ++j) {   // hidden units; LDS broadcast reads
            float wa = lw[j], wb = lw[32 + j], wc = lw[64 + j];
            float bj = lw[96 + j], w2 = lw[128 + j];
            #pragma unroll
            for (int t = 0; t < 8; ++t) {
                float v = fmaf(dst[t], wa, fmaf(dla[t], wb, fmaf(dlo[t], wc, bj)));
                float h = 0.5f * v * (1.0f + erff(v * 0.70710678118654752f));
                acc[t] = fmaf(h, w2, acc[t]);
            }
        }
        #pragma unroll
        for (int t = 0; t < 8; ++t) {
            float x = acc[t];
            float sp = fmaxf(x, 0.0f) + log1pf(expf(-fabsf(x)));  // stable softplus
            rall[cb * 8 + t] = sp * msk[t];
        }
    }

    // softmax over the 1024 stations (per-lane then wave32 reduction)
    float m = rall[0];
    #pragma unroll
    for (int i = 1; i < 32; ++i) m = fmaxf(m, rall[i]);
    #pragma unroll
    for (int off = 16; off >= 1; off >>= 1) m = fmaxf(m, __shfl_xor(m, off, 32));

    float sum = 0.0f;
    #pragma unroll
    for (int i = 0; i < 32; ++i) sum += expf(rall[i] - m);
    #pragma unroll
    for (int off = 16; off >= 1; off >>= 1) sum += __shfl_xor(sum, off, 32);

    float inv = 1.0f / sum;
    half_t* Prow = P + (size_t)(b * GRID_PTS + g) * NST;
    #pragma unroll
    for (int i = 0; i < 32; ++i)
        Prow[(i << 5) + lane] = (half_t)(expf(rall[i] - m) * inv);  // coalesced
}

// ---------------------------------------------------------------------------
// Kernel 2: GEMM out[b][d][g] = sum_s P[b][g][s] * Ft[b][d][s]
// One block (4 waves) per (b, g-tile). The 16x1024 f16 A stripe is DMA'd into
// LDS once via the Tensor Data Mover (with 16B padding per 1KB row-half for
// bank rotation), then each wave computes 4 d-tiles: 4 WMMAs per K-chunk.
// ---------------------------------------------------------------------------
#define TDM_PAD_LDS(off) ((off) + (((off) >> 10) << 4))   // +16B per 1024B
#define A_LDS_BYTES (16 * NST * 2 + (16 * NST * 2 / 1024) * 16)  // 32KB + 512B

__global__ void wmma_gemm_kernel(const half_t* __restrict__ P,
                                 const half_t* __restrict__ Ft,
                                 float* __restrict__ out) {
    __shared__ __attribute__((aligned(16))) unsigned char smemA[A_LDS_BYTES];

    const int lane = threadIdx.x & 31;
    const int wid  = threadIdx.x >> 5;          // 0..3
    const int ntG  = GRID_PTS / 16;             // 512
    const int b    = blockIdx.x / ntG;
    const int gt   = blockIdx.x % ntG;

    const half_t* Abase = P + (size_t)(b * GRID_PTS + gt * 16) * NST; // 32KB stripe

    if (wid == 0) {
        // ---- Tensor DMA descriptor (D#) per CDNA5 ISA ch.8 ----
        unsigned long long ga = (unsigned long long)(uintptr_t)Abase;
        unsigned lds_base = (unsigned)(uintptr_t)(void*)&smemA[0]; // LDS byte offset

        v4u g0;
        g0[0] = 1u;                              // count=1, user mode
        g0[1] = lds_base;                        // lds_addr
        g0[2] = (unsigned)(ga & 0xFFFFFFFFu);    // global_addr[31:0]
        g0[3] = (unsigned)((ga >> 32) & 0x1FFFFFFu) | (2u << 30); // addr[56:32], type=2

        v8u g1;
        g1[0] = (1u << 16)                       // data_size = 2 bytes
              | (1u << 20)                       // pad_enable
              | (7u << 22)                       // pad_interval: 256 DWORDs (1KB)
              | (3u << 25);                      // pad_amount: 4 DWORDs (16B)
        g1[1] = ((unsigned)NST & 0xFFFFu) << 16; // tensor_dim0[15:0] = 1024
        g1[2] = (16u << 16);                     // tensor_dim0[31:16]=0, tensor_dim1=16
        g1[3] = ((unsigned)NST & 0xFFFFu) << 16; // tensor_dim1 hi=0, tile_dim0 = 1024
        g1[4] = 16u;                             // tile_dim1 = 16, tile_dim2 = 0
        g1[5] = (unsigned)NST;                   // tensor_dim0_stride = 1024 elems
        g1[6] = 0u;                              // stride0 hi, tensor_dim1_stride lo
        g1[7] = 0u;                              // tensor_dim1_stride hi
        v4u g2 = {0u, 0u, 0u, 0u};               // dims 2/3 unused (tile dims = 0)
        v4u g3 = {0u, 0u, 0u, 0u};

        asm volatile("tensor_load_to_lds %0, %1, %2, %3"
                     :: "s"(g0), "s"(g1), "s"(g2), "s"(g3)
                     : "memory");
        __builtin_amdgcn_s_wait_tensorcnt((unsigned short)0);
    }
    __syncthreads();

    const int row = lane & 15;
    const int hi  = lane >> 4;

    // B layout used here (contiguous k, same as A): lanes<16 hold N=row with K=0..15; lanes>=16 K=16..31
    const size_t bRow = (size_t)row * NST + (size_t)hi * 16;
    const half_t* Bb[4];
    #pragma unroll
    for (int i = 0; i < 4; ++i)
        Bb[i] = Ft + (size_t)(b * HIDDEN_DIM + (wid * 4 + i) * 16) * NST + bRow;

    // A layout: lanes<16 hold M=row, K=0..15; lanes>=16 K=16..31
    const unsigned aByte = (unsigned)(row * NST + hi * 16) * 2u;

    v8f c0 = {}, c1 = {}, c2 = {}, c3 = {};
    #pragma unroll 2
    for (int kc = 0; kc < NST / 32; ++kc) {
        const int k0 = kc * 32;
        unsigned b0 = aByte + (unsigned)k0 * 2u;        // halves [0..7]  (+hi*16)
        unsigned b1 = b0 + 16u;                         // halves [8..15] (+hi*16)
        union { v16h h; v4f f[2]; } a;
        a.f[0] = *(const v4f*)(smemA + TDM_PAD_LDS(b0));
        a.f[1] = *(const v4f*)(smemA + TDM_PAD_LDS(b1));

        union { v16h h; v4f f[2]; } bm0, bm1, bm2, bm3;
        bm0.f[0] = *(const v4f*)(Bb[0] + k0);
        bm0.f[1] = *(const v4f*)(Bb[0] + k0 + 8);
        bm1.f[0] = *(const v4f*)(Bb[1] + k0);
        bm1.f[1] = *(const v4f*)(Bb[1] + k0 + 8);
        bm2.f[0] = *(const v4f*)(Bb[2] + k0);
        bm2.f[1] = *(const v4f*)(Bb[2] + k0 + 8);
        bm3.f[0] = *(const v4f*)(Bb[3] + k0);
        bm3.f[1] = *(const v4f*)(Bb[3] + k0 + 8);
        __builtin_prefetch(Bb[0] + k0 + 32, 0, 0);
        __builtin_prefetch(Bb[2] + k0 + 32, 0, 0);

        c0 = __builtin_amdgcn_wmma_f32_16x16x32_f16(false, a.h, false, bm0.h,
                                                    (short)0, c0, false, false);
        c1 = __builtin_amdgcn_wmma_f32_16x16x32_f16(false, a.h, false, bm1.h,
                                                    (short)0, c1, false, false);
        c2 = __builtin_amdgcn_wmma_f32_16x16x32_f16(false, a.h, false, bm2.h,
                                                    (short)0, c2, false, false);
        c3 = __builtin_amdgcn_wmma_f32_16x16x32_f16(false, a.h, false, bm3.h,
                                                    (short)0, c3, false, false);
    }

    // C layout: VGPR r, lanes 0-15 -> (M=r, N=lane); lanes 16-31 -> (M=8+r, N-16)
    // out[b][d][g]: d = dtile*16 + row (N), g = gt*16 + hi*8 + r (M)
    v8f cc[4] = {c0, c1, c2, c3};
    #pragma unroll
    for (int i = 0; i < 4; ++i) {
        float* obase = out + ((size_t)(b * HIDDEN_DIM + (wid * 4 + i) * 16 + row))
                             * GRID_PTS + gt * 16 + hi * 8;
        v4f s0 = { cc[i][0], cc[i][1], cc[i][2], cc[i][3] };
        v4f s1 = { cc[i][4], cc[i][5], cc[i][6], cc[i][7] };
        *(v4f*)(obase)     = s0;
        *(v4f*)(obase + 4) = s1;
    }
}

// ---------------------------------------------------------------------------
extern "C" void kernel_launch(void* const* d_in, const int* in_sizes, int n_in,
                              void* d_out, int out_size, void* d_ws, size_t ws_size,
                              hipStream_t stream) {
    const float* feat   = (const float*)d_in[0];  // [B,S,D]
    const float* coords = (const float*)d_in[1];  // [B,S,3]
    const float* mask   = (const float*)d_in[2];  // [B,S]
    const float* W1     = (const float*)d_in[3];  // [3,32]
    const float* b1     = (const float*)d_in[4];  // [32]
    const float* W2     = (const float*)d_in[5];  // [32,1]
    const float* b2     = (const float*)d_in[6];  // [1]
    float* out = (float*)d_out;                   // [B,D,NLAT,NLON]

    // workspace: P (f16, 32 MB) then Ft (f16, 1 MB)
    half_t* P  = (half_t*)d_ws;
    half_t* Ft = (half_t*)((char*)d_ws + (size_t)BATCH * GRID_PTS * NST * sizeof(half_t));

    {
        int total = BATCH * NST * HIDDEN_DIM;  // 524288
        feat_transpose_kernel<<<(total + 255) / 256, 256, 0, stream>>>(feat, Ft);
    }
    {
        int waves = BATCH * GRID_PTS;          // 16384 waves, 4 per block
        station_weights_kernel<<<waves / 4, 128, 0, stream>>>(coords, mask, W1, b1,
                                                              W2, b2, P);
    }
    {
        int blocks = BATCH * (GRID_PTS / 16);  // 1024 blocks, 4 waves each
        wmma_gemm_kernel<<<blocks, 128, 0, stream>>>(P, Ft, out);
    }
}
